// SpikingNeuronLayer_76673756168501
// MI455X (gfx1250) — matrix operand order repaired and checked by reading, not compiled, not run on hardware
//
#include <hip/hip_runtime.h>

// Spiking neuron layer, fused bf16-split WMMA GEMM + LIF scan for gfx1250.
// B=64, T=512, IN=1024, HID=1024. fp32 accuracy via 3-term split-bf16 WMMA.

#define B_    64
#define T_    512
#define IN_   1024
#define HID_  1024
#define ALPHA_ 0.9512294245007140f   // exp(-1/20)

typedef __attribute__((ext_vector_type(16))) __bf16       v16bf;
typedef __attribute__((ext_vector_type(8)))  float        v8f;
typedef __attribute__((ext_vector_type(4)))  float        v4f;
typedef __attribute__((ext_vector_type(4)))  unsigned int v4u;
typedef __attribute__((ext_vector_type(2)))  unsigned int v2u;

union Frag { v16bf bf; v4u q[2]; };

__device__ __forceinline__ unsigned short f2bf(float f) {
    unsigned u = __float_as_uint(f);
    u += 0x7FFFu + ((u >> 16) & 1u);          // round-to-nearest-even
    return (unsigned short)(u >> 16);
}
__device__ __forceinline__ float bf2f(unsigned short h) {
    return __uint_as_float(((unsigned)h) << 16);
}
__device__ __forceinline__ unsigned int pk2(unsigned short a, unsigned short b) {
    return (unsigned)a | ((unsigned)b << 16);
}

#define WMMA_BF16(A, Bm, C) \
    __builtin_amdgcn_wmma_f32_16x16x32_bf16(false, (A).bf, false, (Bm).bf, (short)0, (C), false, false)

// ---------------------------------------------------------------------------
// Prologue: split W (fp32 [HID][IN]) into bf16 hi/lo arrays in workspace.
// ---------------------------------------------------------------------------
__global__ void snn_wconv(const float* __restrict__ W,
                          unsigned short* __restrict__ whi,
                          unsigned short* __restrict__ wlo) {
    const int n4 = (HID_ * IN_) / 4;
    for (int j = blockIdx.x * blockDim.x + threadIdx.x; j < n4;
         j += gridDim.x * blockDim.x) {
        v4f v = ((const v4f*)W)[j];
        unsigned short h0 = f2bf(v.x), h1 = f2bf(v.y), h2 = f2bf(v.z), h3 = f2bf(v.w);
        unsigned short l0 = f2bf(v.x - bf2f(h0)), l1 = f2bf(v.y - bf2f(h1));
        unsigned short l2 = f2bf(v.z - bf2f(h2)), l3 = f2bf(v.w - bf2f(h3));
        v2u hv; hv.x = pk2(h0, h1); hv.y = pk2(h2, h3);
        v2u lv; lv.x = pk2(l0, l1); lv.y = pk2(l2, l3);
        ((v2u*)whi)[j] = hv;
        ((v2u*)wlo)[j] = lv;
    }
}

// ---------------------------------------------------------------------------
// 16-step LIF scan of one 16(t) x 16(h)-per-halfwave tile pair.
// a0 holds columns hwave+0..15, a1 holds columns hwave+16..31 (WMMA D layout:
// lane L<16 has rows 0..7 of col L, lane L+16 has rows 8..15 of col L).
// SWAPX16 swizzle gives every lane the full 16-row column it owns.
// ---------------------------------------------------------------------------
__device__ __forceinline__ void scan16(const v8f& a0, const v8f& a1, bool loHalf,
                                       float bcol, float& mem,
                                       float* __restrict__ spk,
                                       int tbase, int hcol) {
    float col[16];
#pragma unroll
    for (int r = 0; r < 8; ++r) {
        float send = loHalf ? a1[r] : a0[r];
        float recv = __uint_as_float(
            (unsigned)__builtin_amdgcn_ds_swizzle(__float_as_int(send), 0x401F));
        col[r]     = loHalf ? a0[r] : recv;
        col[8 + r] = loHalf ? recv  : a1[r];
    }
#pragma unroll
    for (int r = 0; r < 16; ++r) {
        mem = ALPHA_ * mem + (col[r] + bcol);
        bool fire = (mem >= 1.0f);
        spk[(size_t)(tbase + r) * HID_ + hcol] = fire ? 1.0f : 0.0f;
        mem = fire ? 0.0f : mem;
    }
}

// ---------------------------------------------------------------------------
// Fused kernel: grid (HID/256, B), 256 threads = 8 waves.
// Wave owns 32 neuron columns; block walks T in chunks of 32 timesteps.
// ---------------------------------------------------------------------------
#define XSTRIDE 1032   // 1024 + 8 ushort pad (16B) to break LDS bank conflicts

__global__ __launch_bounds__(256)
void snn_fused(const float* __restrict__ x, const float* __restrict__ bias,
               const unsigned short* __restrict__ whi,
               const unsigned short* __restrict__ wlo,
               float* __restrict__ spikes, float* __restrict__ memout) {
    __shared__ unsigned short xs_hi[32 * XSTRIDE];
    __shared__ unsigned short xs_lo[32 * XSTRIDE];

    const int  lane   = threadIdx.x & 31;
    const int  wv     = threadIdx.x >> 5;
    const int  b      = blockIdx.y;
    const int  hwave  = blockIdx.x * 256 + wv * 32;
    const int  hcol   = hwave + lane;
    const int  ln15   = lane & 15;
    const bool loHalf = (lane < 16);

    const float* xb  = x + (size_t)b * T_ * IN_;
    float*       spk = spikes + (size_t)b * T_ * HID_;
    const float  bcol = bias[hcol];
    float        mem  = 0.0f;

    // B-fragment row pointers (W row = neuron column), split K halves per lane.
    const int bkoff = loHalf ? 0 : 16;
    const unsigned short* w0hp = whi + (size_t)(hwave + ln15)      * IN_ + bkoff;
    const unsigned short* w0lp = wlo + (size_t)(hwave + ln15)      * IN_ + bkoff;
    const unsigned short* w1hp = whi + (size_t)(hwave + 16 + ln15) * IN_ + bkoff;
    const unsigned short* w1lp = wlo + (size_t)(hwave + 16 + ln15) * IN_ + bkoff;

    const int abase = loHalf ? 0 : 16;  // A-frag K sub-base within 32-chunk

    for (int tc = 0; tc < T_ / 32; ++tc) {
        const int t0 = tc * 32;

        // Stage x[b, t0..t0+31, :] into LDS as split bf16 (shared by 8 waves).
        for (int f4 = threadIdx.x; f4 < 32 * 256; f4 += 256) {
            const int r = f4 >> 8;
            const int k = (f4 & 255) << 2;
            v4f v = *(const v4f*)(xb + (size_t)(t0 + r) * IN_ + k);
            unsigned short h0 = f2bf(v.x), h1 = f2bf(v.y), h2 = f2bf(v.z), h3 = f2bf(v.w);
            unsigned short l0 = f2bf(v.x - bf2f(h0)), l1 = f2bf(v.y - bf2f(h1));
            unsigned short l2 = f2bf(v.z - bf2f(h2)), l3 = f2bf(v.w - bf2f(h3));
            v2u hv; hv.x = pk2(h0, h1); hv.y = pk2(h2, h3);
            v2u lv; lv.x = pk2(l0, l1); lv.y = pk2(l2, l3);
            *(v2u*)(&xs_hi[r * XSTRIDE + k]) = hv;
            *(v2u*)(&xs_lo[r * XSTRIDE + k]) = lv;
        }
        __syncthreads();

        v8f acc00 = {0,0,0,0,0,0,0,0}, acc01 = {0,0,0,0,0,0,0,0};
        v8f acc10 = {0,0,0,0,0,0,0,0}, acc11 = {0,0,0,0,0,0,0,0};

        for (int k0 = 0; k0 < IN_; k0 += 32) {
            Frag b0h, b0l, b1h, b1l;
            b0h.q[0] = *(const v4u*)(w0hp + k0); b0h.q[1] = *(const v4u*)(w0hp + k0 + 8);
            b0l.q[0] = *(const v4u*)(w0lp + k0); b0l.q[1] = *(const v4u*)(w0lp + k0 + 8);
            b1h.q[0] = *(const v4u*)(w1hp + k0); b1h.q[1] = *(const v4u*)(w1hp + k0 + 8);
            b1l.q[0] = *(const v4u*)(w1lp + k0); b1l.q[1] = *(const v4u*)(w1lp + k0 + 8);

            {   // M-tile 0: timesteps t0..t0+15
                const int ao = ln15 * XSTRIDE + abase + k0;
                Frag ah, al;
                ah.q[0] = *(const v4u*)(&xs_hi[ao]); ah.q[1] = *(const v4u*)(&xs_hi[ao + 8]);
                al.q[0] = *(const v4u*)(&xs_lo[ao]); al.q[1] = *(const v4u*)(&xs_lo[ao + 8]);
                acc00 = WMMA_BF16(ah, b0h, acc00);
                acc00 = WMMA_BF16(ah, b0l, acc00);
                acc00 = WMMA_BF16(al, b0h, acc00);
                acc01 = WMMA_BF16(ah, b1h, acc01);
                acc01 = WMMA_BF16(ah, b1l, acc01);
                acc01 = WMMA_BF16(al, b1h, acc01);
            }
            {   // M-tile 1: timesteps t0+16..t0+31
                const int ao = (16 + ln15) * XSTRIDE + abase + k0;
                Frag ah, al;
                ah.q[0] = *(const v4u*)(&xs_hi[ao]); ah.q[1] = *(const v4u*)(&xs_hi[ao + 8]);
                al.q[0] = *(const v4u*)(&xs_lo[ao]); al.q[1] = *(const v4u*)(&xs_lo[ao + 8]);
                acc10 = WMMA_BF16(ah, b0h, acc10);
                acc10 = WMMA_BF16(ah, b0l, acc10);
                acc10 = WMMA_BF16(al, b0h, acc10);
                acc11 = WMMA_BF16(ah, b1h, acc11);
                acc11 = WMMA_BF16(ah, b1l, acc11);
                acc11 = WMMA_BF16(al, b1h, acc11);
            }
        }

        // Sequential LIF scan of the 32 timesteps (mem carried in register).
        scan16(acc00, acc01, loHalf, bcol, mem, spk, t0,      hcol);
        scan16(acc10, acc11, loHalf, bcol, mem, spk, t0 + 16, hcol);

        __syncthreads();   // protect LDS before next chunk's staging
    }

    memout[(size_t)b * HID_ + hcol] = mem;
}

extern "C" void kernel_launch(void* const* d_in, const int* in_sizes, int n_in,
                              void* d_out, int out_size, void* d_ws, size_t ws_size,
                              hipStream_t stream) {
    const float* x    = (const float*)d_in[0];   // [B,T,IN] f32
    const float* W    = (const float*)d_in[1];   // [HID,IN] f32
    const float* bias = (const float*)d_in[2];   // [HID]    f32
    float* out = (float*)d_out;                  // spikes [B,T,HID] ++ mem_final [B,HID]

    unsigned short* whi = (unsigned short*)d_ws;               // 2 MB
    unsigned short* wlo = whi + (size_t)HID_ * IN_;            // 2 MB

    snn_wconv<<<1024, 256, 0, stream>>>(W, whi, wlo);

    dim3 grid(HID_ / 256, B_);
    snn_fused<<<grid, 256, 0, stream>>>(x, bias, whi, wlo,
                                        out, out + (size_t)B_ * T_ * HID_);
}
